// MRGPTQLinear_5257039970429
// MI455X (gfx1250) — compile-verified, matrix operand order checked by reading, not run on hardware
//
#include <hip/hip_runtime.h>
#include <hip/hip_bf16.h>

typedef __attribute__((ext_vector_type(16))) _Float16 v16h;
typedef __attribute__((ext_vector_type(8)))  _Float16 v8h;
typedef __attribute__((ext_vector_type(8)))  float    v8f;

#define M_ROWS 16384
#define K_DIM  1024
#define N_DIM  1024
#define R_DIM  32

// ---------------------------------------------------------------------------
// Stage 1: f32 -> f16 conversion (weights, lora_b)
// ---------------------------------------------------------------------------
__global__ void f32_to_f16_kernel(const float* __restrict__ in,
                                  _Float16* __restrict__ out, int n) {
    int i = blockIdx.x * blockDim.x + threadIdx.x;
    if (i < n) out[i] = (_Float16)in[i];
}

// ---------------------------------------------------------------------------
// Stage 2: per-row prep. One block (64 threads = 2 waves) per row of x.
//   - act_order gather via perm
//   - H16 rotation per 16-block (H16[k][i] = parity(popc(k&i)) ? -0.25 : 0.25)
//   - NVFP4 block-16 quantization (scale = amax/6, nearest of 8 levels,
//     midpoint ties resolve to the LOWER level, matching jnp.argmin)
//   - t = x_rot @ lora_a^T  (per-row R=32 dot products), stored f16
// ---------------------------------------------------------------------------
__global__ void prep_kernel(const float* __restrict__ x,
                            const int*  __restrict__ perm,
                            const float* __restrict__ lora_a,
                            _Float16* __restrict__ xq,
                            _Float16* __restrict__ t16) {
    __shared__ float xrot_s[K_DIM];

    const int row = blockIdx.x;          // 0 .. M_ROWS-1
    const int tid = threadIdx.x;         // 0 .. 63
    const float* __restrict__ xrow = x + (size_t)row * K_DIM;

    // Each thread owns one 16-element block of this row.
    const int blk = tid;                 // 0 .. 63
    float xp[16];
#pragma unroll
    for (int i = 0; i < 16; ++i)
        xp[i] = xrow[perm[blk * 16 + i]];

    // Hadamard-16 (symmetric): y[i] = sum_k xp[k] * (+-0.25)
    float y[16];
#pragma unroll
    for (int i = 0; i < 16; ++i) {
        float s = 0.0f;
#pragma unroll
        for (int k = 0; k < 16; ++k) {
            const float hv = (__popc(i & k) & 1) ? -0.25f : 0.25f;
            s = fmaf(xp[k], hv, s);
        }
        y[i] = s;
    }

    // amax (clipped at 1e-12), scale = amax/6
    float amax = 1e-12f;
#pragma unroll
    for (int i = 0; i < 16; ++i) amax = fmaxf(amax, fabsf(y[i]));
    const float scale = amax * (1.0f / 6.0f);
    const float inv_s = 6.0f / amax;

#pragma unroll
    for (int i = 0; i < 16; ++i) {
        const float an = fabsf(y[i]) * inv_s;
        // NVFP4 levels {0,.5,1,1.5,2,3,4,6}; midpoints 0.25,.75,1.25,1.75,2.5,3.5,5
        float lev;
        if      (an <= 0.25f) lev = 0.0f;
        else if (an <= 0.75f) lev = 0.5f;
        else if (an <= 1.25f) lev = 1.0f;
        else if (an <= 1.75f) lev = 1.5f;
        else if (an <= 2.5f)  lev = 2.0f;
        else if (an <= 3.5f)  lev = 3.0f;
        else if (an <= 5.0f)  lev = 4.0f;
        else                  lev = 6.0f;
        const float q = copysignf(lev * scale, y[i]);
        xq[(size_t)row * K_DIM + blk * 16 + i] = (_Float16)q;
        xrot_s[blk * 16 + i] = y[i];
    }
    __syncthreads();

    // t[row][r] = dot(x_rot_row, lora_a[r]) for r = 0..31 (threads 0..31)
    if (tid < R_DIM) {
        const float4* __restrict__ ar4 = (const float4*)(lora_a + (size_t)tid * K_DIM);
        const float4* __restrict__ xs4 = (const float4*)xrot_s;
        float s = 0.0f;
        for (int k = 0; k < K_DIM / 4; ++k) {
            const float4 a4 = ar4[k];
            const float4 x4 = xs4[k];
            s = fmaf(a4.x, x4.x, s);
            s = fmaf(a4.y, x4.y, s);
            s = fmaf(a4.z, x4.z, s);
            s = fmaf(a4.w, x4.w, s);
        }
        t16[(size_t)row * R_DIM + tid] = (_Float16)s;
    }
}

// ---------------------------------------------------------------------------
// Stage 3: WMMA GEMM  out = xq @ w16^T  (+ t16 @ lb16^T + bias)
// 256 threads = 8 waves/block arranged 2(M) x 4(N); each wave computes a
// 64x32 output macro-tile as 4x2 WMMA tiles (register blocking): per K-step
// it loads 6 fragments (12x b128) and issues 8 v_wmma_f32_16x16x32_f16 ->
// 0.67 WMMA per vector load (A frags reused 2x, B frags reused 4x).
//
// __launch_bounds__(256, 1): allow the full 256-VGPR budget (one wave/EU
// min) so the 8 f32 accumulator tiles + double-buffered fragments stay in
// registers with no scratch spills. Row addressing uses two base pointers
// plus compile-time offsets (i*16*K rows = 32768*i bytes, inside the signed
// 24-bit global IOFFSET) instead of per-row pointer arrays, freeing VGPRs.
//
// Fragment layout (ISA 7.12.2, 16-bit A 16x32): lane L -> row M = L%16,
// half h = L/16; VGPRs 0..3 hold K = h*8+{0..7}, VGPRs 4..7 hold K =
// 16+h*8+{0..7} -> two contiguous 16B chunks per lane. w_q is (N,K)
// row-major = column-major B tile -> same per-lane pattern over row n.
// C/D: VGPR j holds (m = tile_m + h*8 + j, n = tile_n + L%16).
// ---------------------------------------------------------------------------
__global__ __launch_bounds__(256, 1)
void gemm_kernel(const _Float16* __restrict__ xq,
                 const _Float16* __restrict__ w16,
                 const _Float16* __restrict__ t16,
                 const _Float16* __restrict__ lb16,
                 const float* __restrict__ bias,
                 float* __restrict__ out) {
    const int wave = threadIdx.x >> 5;
    const int lane = threadIdx.x & 31;
    const int r    = lane & 15;        // row within A tile / col within B tile
    const int hh   = lane >> 4;        // K-half selector

    const int wm = wave >> 2;          // wave position along M (0..1)
    const int wn = wave & 3;           // wave position along N (0..3)

    const int m_base = blockIdx.x * 128 + wm * 64;   // 64 rows per wave
    const int n_base = blockIdx.y * 128 + wn * 32;   // 32 cols per wave

    // Single base pointer per matrix; tile rows addressed via constant offsets.
    const _Float16* __restrict__ Abase = xq  + (size_t)(m_base + r) * K_DIM + hh * 8;
    const _Float16* __restrict__ Bbase = w16 + (size_t)(n_base + r) * K_DIM + hh * 8;

    v8f acc[4][2];
#pragma unroll
    for (int i = 0; i < 4; ++i)
#pragma unroll
        for (int j = 0; j < 2; ++j)
            acc[i][j] = (v8f){};

    union Frag { v16h v; v8h h[2]; };

    for (int k0 = 0; k0 < K_DIM; k0 += 32) {
        Frag a[4], b[2];
#pragma unroll
        for (int i = 0; i < 4; ++i) {
            a[i].h[0] = *(const v8h*)(Abase + (size_t)i * 16 * K_DIM + k0);
            a[i].h[1] = *(const v8h*)(Abase + (size_t)i * 16 * K_DIM + k0 + 16);
        }
#pragma unroll
        for (int j = 0; j < 2; ++j) {
            b[j].h[0] = *(const v8h*)(Bbase + (size_t)j * 16 * K_DIM + k0);
            b[j].h[1] = *(const v8h*)(Bbase + (size_t)j * 16 * K_DIM + k0 + 16);
        }
#pragma unroll
        for (int i = 0; i < 4; ++i)
#pragma unroll
            for (int j = 0; j < 2; ++j)
                acc[i][j] = __builtin_amdgcn_wmma_f32_16x16x32_f16(
                    false, a[i].v, false, b[j].v, (short)0, acc[i][j], false, false);
    }

    // Fused LoRA epilogue: t (Mx32) @ lora_b^T (Nx32) = 8 more 16x16x32 WMMAs
    {
        const _Float16* __restrict__ Tbase = t16  + (size_t)(m_base + r) * R_DIM + hh * 8;
        const _Float16* __restrict__ Lbase = lb16 + (size_t)(n_base + r) * R_DIM + hh * 8;
        Frag ta[4], tb[2];
#pragma unroll
        for (int i = 0; i < 4; ++i) {
            ta[i].h[0] = *(const v8h*)(Tbase + (size_t)i * 16 * R_DIM);
            ta[i].h[1] = *(const v8h*)(Tbase + (size_t)i * 16 * R_DIM + 16);
        }
#pragma unroll
        for (int j = 0; j < 2; ++j) {
            tb[j].h[0] = *(const v8h*)(Lbase + (size_t)j * 16 * R_DIM);
            tb[j].h[1] = *(const v8h*)(Lbase + (size_t)j * 16 * R_DIM + 16);
        }
#pragma unroll
        for (int i = 0; i < 4; ++i)
#pragma unroll
            for (int j = 0; j < 2; ++j)
                acc[i][j] = __builtin_amdgcn_wmma_f32_16x16x32_f16(
                    false, ta[i].v, false, tb[j].v, (short)0, acc[i][j], false, false);
    }

    // Store with bias. C/D layout: VGPR j -> m = tile_m + hh*8 + j, n = tile_n + r.
#pragma unroll
    for (int jn = 0; jn < 2; ++jn) {
        const int n = n_base + 16 * jn + r;
        const float bv = bias[n];
#pragma unroll
        for (int im = 0; im < 4; ++im) {
            const int m0 = m_base + 16 * im + hh * 8;
#pragma unroll
            for (int j = 0; j < 8; ++j) {
                out[(size_t)(m0 + j) * N_DIM + n] = acc[im][jn][j] + bv;
            }
        }
    }
}

// ---------------------------------------------------------------------------
// Launch
// ---------------------------------------------------------------------------
extern "C" void kernel_launch(void* const* d_in, const int* in_sizes, int n_in,
                              void* d_out, int out_size, void* d_ws, size_t ws_size,
                              hipStream_t stream) {
    const float* x      = (const float*)d_in[0];   // [2,8192,1024]
    const float* w_q    = (const float*)d_in[1];   // [1024,1024]
    const float* lora_a = (const float*)d_in[2];   // [32,1024]
    const float* lora_b = (const float*)d_in[3];   // [1024,32]
    const float* bias   = (const float*)d_in[4];   // [1024]
    // d_in[5] = H16 (recomputed analytically), d_in[6] = perm
    const int*   perm   = (const int*)d_in[6];     // [1024]

    float* out = (float*)d_out;

    // Workspace layout
    char* ws = (char*)d_ws;
    _Float16* xq   = (_Float16*)(ws);                                   // 32 MB
    _Float16* w16  = (_Float16*)(ws + (size_t)M_ROWS * K_DIM * 2);      //  2 MB
    _Float16* t16  = (_Float16*)(ws + (size_t)M_ROWS * K_DIM * 2
                                    + (size_t)N_DIM * K_DIM * 2);       //  1 MB
    _Float16* lb16 = (_Float16*)(ws + (size_t)M_ROWS * K_DIM * 2
                                    + (size_t)N_DIM * K_DIM * 2
                                    + (size_t)M_ROWS * R_DIM * 2);      // 64 KB

    // Stage 1: convert weights + lora_b to f16
    {
        int n = N_DIM * K_DIM;
        f32_to_f16_kernel<<<(n + 255) / 256, 256, 0, stream>>>(w_q, w16, n);
        int n2 = N_DIM * R_DIM;
        f32_to_f16_kernel<<<(n2 + 255) / 256, 256, 0, stream>>>(lora_b, lb16, n2);
    }

    // Stage 2: gather + H16 + NVFP4 quant + LoRA-A projection
    prep_kernel<<<M_ROWS, 64, 0, stream>>>(x, perm, lora_a, xq, t16);

    // Stage 3: WMMA GEMM with fused LoRA + bias (4x2 register-blocked waves)
    dim3 grid(M_ROWS / 128, N_DIM / 128);   // (128, 8)
    gemm_kernel<<<grid, 256, 0, stream>>>(xq, w16, t16, lb16, bias, out);
}
